// MultiScalePointCloudEncoder_70179765617261
// MI455X (gfx1250) — compile-verified, matrix operand order checked by reading, not run on hardware
//
#include <hip/hip_runtime.h>
#include <hip/hip_bf16.h>

typedef __attribute__((ext_vector_type(16))) __bf16 v16bf;
typedef __attribute__((ext_vector_type(8)))  float  v8f;

#define NPTS 8192
#define DXP  19
#define KN1  20
#define KN2  40
#define NBR2 41
#define KN3  80
#define H1D  128
#define H2D  256
#define H3D  512
#define OUTD 1024
#define CHNK 1024
#define E1   (NPTS*KN1)
#define E2   (CHNK*NBR2)
#define EPAD 64     // edge feature width 38 padded to k-step multiple
#define RPAD 32     // rel width 3 padded
#define KSUB 256    // K elements staged in LDS per super-tile (32KB for 4 N-tiles)
#define KSUBT (KSUB/32)

// ---------- helpers ----------
__device__ __forceinline__ unsigned short f2bf(float f){
  unsigned u = __builtin_bit_cast(unsigned, f);
  u += 0x7FFFu + ((u >> 16) & 1u);   // round-to-nearest-even bf16
  return (unsigned short)(u >> 16);
}
__device__ __forceinline__ float gelu_exact(float x){
  return 0.5f * x * (1.f + erff(x * 0.70710678118654752f));
}
__device__ __forceinline__ unsigned long long ullmin2(unsigned long long a, unsigned long long b){
  return a < b ? a : b;
}

struct Frag32B { uint4 lo, hi; };
// A-fragment: lane holds row (lane&15); k = k0 + half*8 + {0..7} and k0 + 16 + half*8 + {0..7}
__device__ __forceinline__ v16bf load_afrag(const unsigned short* p){
  Frag32B f; f.lo = *(const uint4*)p; f.hi = *(const uint4*)(p + 16);
  return __builtin_bit_cast(v16bf, f);
}
// B-fragment: pre-packed contiguous 16 bf16 per lane (works for global or LDS pointers)
__device__ __forceinline__ v16bf load_bfrag(const unsigned short* p){
  Frag32B f; f.lo = *(const uint4*)p; f.hi = *(const uint4*)(p + 8);
  return __builtin_bit_cast(v16bf, f);
}

// ---------- weight pack: f32 [K x N] -> bf16 WMMA B-fragments, K zero-padded to Kp ----------
// packed element t = ((nt*nKt + kt)*32 + lane)*16 + e
__global__ void pack_w_kernel(const float* __restrict__ W, unsigned short* __restrict__ Wp,
                              int N, int K, int Kp){
  int t = blockIdx.x * blockDim.x + threadIdx.x;
  if (t >= N * Kp) return;
  int e    = t & 15;
  int lane = (t >> 4) & 31;
  int rest = t >> 9;
  int nKt  = Kp >> 5;
  int kt   = rest % nKt;
  int nt   = rest / nKt;
  int half = lane >> 4;
  int v    = e >> 1;
  int kb   = (v < 4) ? (half * 8 + v * 2) : (16 + half * 8 + (v - 4) * 2);
  int k    = kt * 32 + kb + (e & 1);
  int nc   = nt * 16 + (lane & 15);
  float val = (k < K) ? W[(size_t)k * N + nc] : 0.f;
  Wp[t] = f2bf(val);
}

// ---------- WMMA GEMM: block = 4 waves = 64x64 tile; B staged in LDS ----------
// All 4 waves share nt0 -> the packed B super-tile is staged into LDS once per block
// (coalesced b128 copies), then read as ds_load_b128 fragments. A streams from global.
// A: [M x Kp] bf16 (Kp multiple of 32, zero padded). Wp: packed fragments.
// Cf (f32) and/or Cb (bf16) outputs, both stride N. act: 0=none, 1=relu.
__global__ void __launch_bounds__(128)
wmma_gemm_kernel(const unsigned short* __restrict__ A,
                 const unsigned short* __restrict__ Wp,
                 const float* __restrict__ bias,
                 float* __restrict__ Cf,
                 unsigned short* __restrict__ Cb,
                 int M, int N, int Kp, int act)
{
  __shared__ unsigned short bsm[4 * KSUBT * 512];   // 32 KB
  const int lane = threadIdx.x & 31;
  const int wave = threadIdx.x >> 5;
  const int mt = blockIdx.x * 4 + wave;
  const int mtTiles = M >> 4;
  const int mtc = (mt < mtTiles) ? mt : (mtTiles - 1);   // clamp; stores predicated below
  const int nt0  = blockIdx.y * 4;
  const int half = lane >> 4;
  const int l15  = lane & 15;
  const int nKt  = Kp >> 5;

  const unsigned short* ap = A + (size_t)(mtc * 16 + l15) * Kp + half * 8;
  v8f acc0 = {}, acc1 = {}, acc2 = {}, acc3 = {};

  const int nStage = (Kp + KSUB - 1) / KSUB;
  for (int st = 0; st < nStage; ++st) {
    const int k0  = st * KSUB;
    const int rem = Kp - k0;
    const int ktn = (rem < KSUB ? rem : KSUB) >> 5;   // kt tiles this stage
    __syncthreads();                                   // previous-stage reads done
    // cooperative stage of 4 N-tiles' fragments (contiguous in packed layout)
    for (int t = 0; t < 4; ++t) {
      const uint4* src = (const uint4*)(Wp + ((size_t)(nt0 + t) * nKt + (k0 >> 5)) * 512);
      uint4*       dst = (uint4*)(bsm + t * KSUBT * 512);
      const int nvec = ktn * 64;                       // ktn*512 shorts / 8
      for (int idx = threadIdx.x; idx < nvec; idx += 128) dst[idx] = src[idx];
    }
    __syncthreads();
    const unsigned short* ls = bsm + lane * 16;
    if (ktn == KSUBT) {
#pragma unroll
      for (int kt = 0; kt < KSUBT; ++kt) {
        v16bf a  = load_afrag(ap); ap += 32;
        v16bf b0 = load_bfrag(ls + (0 * KSUBT + kt) * 512);
        v16bf b1 = load_bfrag(ls + (1 * KSUBT + kt) * 512);
        v16bf b2 = load_bfrag(ls + (2 * KSUBT + kt) * 512);
        v16bf b3 = load_bfrag(ls + (3 * KSUBT + kt) * 512);
        acc0 = __builtin_amdgcn_wmma_f32_16x16x32_bf16(false, a, false, b0, (short)0, acc0, false, false);
        acc1 = __builtin_amdgcn_wmma_f32_16x16x32_bf16(false, a, false, b1, (short)0, acc1, false, false);
        acc2 = __builtin_amdgcn_wmma_f32_16x16x32_bf16(false, a, false, b2, (short)0, acc2, false, false);
        acc3 = __builtin_amdgcn_wmma_f32_16x16x32_bf16(false, a, false, b3, (short)0, acc3, false, false);
      }
    } else {
      for (int kt = 0; kt < ktn; ++kt) {
        v16bf a  = load_afrag(ap); ap += 32;
        v16bf b0 = load_bfrag(ls + (0 * KSUBT + kt) * 512);
        v16bf b1 = load_bfrag(ls + (1 * KSUBT + kt) * 512);
        v16bf b2 = load_bfrag(ls + (2 * KSUBT + kt) * 512);
        v16bf b3 = load_bfrag(ls + (3 * KSUBT + kt) * 512);
        acc0 = __builtin_amdgcn_wmma_f32_16x16x32_bf16(false, a, false, b0, (short)0, acc0, false, false);
        acc1 = __builtin_amdgcn_wmma_f32_16x16x32_bf16(false, a, false, b1, (short)0, acc1, false, false);
        acc2 = __builtin_amdgcn_wmma_f32_16x16x32_bf16(false, a, false, b2, (short)0, acc2, false, false);
        acc3 = __builtin_amdgcn_wmma_f32_16x16x32_bf16(false, a, false, b3, (short)0, acc3, false, false);
      }
    }
  }

  if (mt < mtTiles) {                      // wave-uniform store predicate
    const int rowb = mt * 16 + half * 8;   // C/D layout: VGPR r -> row rowb+r, col = nt*16 + l15
#pragma unroll
    for (int t = 0; t < 4; ++t) {
      const int nc = (nt0 + t) * 16 + l15;
      const float bv = bias ? bias[nc] : 0.f;
      v8f acc = (t == 0) ? acc0 : (t == 1) ? acc1 : (t == 2) ? acc2 : acc3;
#pragma unroll
      for (int r = 0; r < 8; ++r) {
        float v = acc[r] + bv;
        if (act == 1) v = fmaxf(v, 0.f);
        const size_t idx = (size_t)(rowb + r) * N + nc;
        if (Cf) Cf[idx] = v;
        if (Cb) Cb[idx] = f2bf(v);
      }
    }
  }
}

// ---------- xp = concat(x, pos) ----------
__global__ void build_xp_kernel(const float* __restrict__ x, const float* __restrict__ pos,
                                float* __restrict__ xp){
  int t = blockIdx.x * blockDim.x + threadIdx.x;
  if (t >= NPTS * DXP) return;
  int i = t / DXP, d = t % DXP;
  xp[t] = (d < 16) ? x[(size_t)i * 16 + d] : pos[(size_t)i * 3 + (d - 16)];
}

// ---------- generic exact kNN: one block (256 thr) per query; iterative min-select ----------
__global__ void __launch_bounds__(256)
knn_kernel(const float* __restrict__ F, int D, int K, int out_stride,
           int exclude_self, int append_self, int* __restrict__ nbout)
{
  __shared__ float d2s[NPTS];
  __shared__ unsigned long long red[256];
  const int i = blockIdx.x;
  float q[DXP];
  for (int d = 0; d < D; ++d) q[d] = F[(size_t)i * D + d];
  for (int j = threadIdx.x; j < NPTS; j += 256) {
    float s = 0.f;
    for (int d = 0; d < D; ++d) { float t = q[d] - F[(size_t)j * D + d]; s += t * t; }
    if (exclude_self && j == i) s = __int_as_float(0x7F800000);
    d2s[j] = fmaxf(s, 0.f);
  }
  __syncthreads();
  unsigned long long prev = 0ull;   // keys (+1)-shifted so 0 == "none yet"
  for (int kk = 0; kk < K; ++kk) {
    unsigned long long best = ~0ull;
    for (int j = threadIdx.x; j < NPTS; j += 256) {
      unsigned int b = __float_as_uint(d2s[j]);
      unsigned long long key = ((((unsigned long long)b) << 13) | (unsigned)j) + 1ull;
      if (key > prev && key < best) best = key;
    }
    red[threadIdx.x] = best;
    __syncthreads();
    for (int s = 128; s > 0; s >>= 1) {
      if (threadIdx.x < s) red[threadIdx.x] = ullmin2(red[threadIdx.x], red[threadIdx.x + s]);
      __syncthreads();
    }
    prev = red[0];
    if (threadIdx.x == 0) nbout[(size_t)i * out_stride + kk] = (int)((prev - 1ull) & 0x1FFFull);
    __syncthreads();
  }
  if (append_self && threadIdx.x == 0) nbout[(size_t)i * out_stride + K] = i;
}

// ---------- stage-1 edge features e = [xi, xj - xi, 0-pad] (bf16, width EPAD) ----------
__global__ void build_edge_kernel(const float* __restrict__ xp, const int* __restrict__ nb1,
                                  unsigned short* __restrict__ e){
  long long t = (long long)blockIdx.x * blockDim.x + threadIdx.x;
  if (t >= (long long)E1 * EPAD) return;
  int edge = (int)(t / EPAD), d = (int)(t % EPAD);
  int i = edge / KN1;
  float v = 0.f;
  if (d < DXP) v = xp[(size_t)i * DXP + d];
  else if (d < 2 * DXP) {
    int j = nb1[edge];
    int dd = d - DXP;
    v = xp[(size_t)j * DXP + dd] - xp[(size_t)i * DXP + dd];
  }
  e[t] = f2bf(v);
}

// ---------- max-pool over K1 neighbors ----------
__global__ void maxpool_kernel(const float* __restrict__ m, float* __restrict__ h1){
  int t = blockIdx.x * blockDim.x + threadIdx.x;
  if (t >= NPTS * H1D) return;
  int i = t / H1D, c = t % H1D;
  float mx = -__int_as_float(0x7F800000);
  for (int j = 0; j < KN1; ++j)
    mx = fmaxf(mx, m[((size_t)i * KN1 + j) * H1D + c]);
  h1[t] = mx;
}

// ---------- LayerNorm + exact GELU (one block per row), optional bf16 copy ----------
__global__ void __launch_bounds__(256)
ln_gelu_kernel(const float* __restrict__ in, const float* __restrict__ g,
               const float* __restrict__ b, float* __restrict__ out,
               unsigned short* __restrict__ outb, int Wd){
  __shared__ float red[256];
  const int i = blockIdx.x;
  const float* row = in + (size_t)i * Wd;
  float s = 0.f;
  for (int c = threadIdx.x; c < Wd; c += 256) s += row[c];
  red[threadIdx.x] = s; __syncthreads();
  for (int t = 128; t > 0; t >>= 1) { if (threadIdx.x < t) red[threadIdx.x] += red[threadIdx.x + t]; __syncthreads(); }
  const float mu = red[0] / Wd;
  __syncthreads();
  float v = 0.f;
  for (int c = threadIdx.x; c < Wd; c += 256) { float d = row[c] - mu; v += d * d; }
  red[threadIdx.x] = v; __syncthreads();
  for (int t = 128; t > 0; t >>= 1) { if (threadIdx.x < t) red[threadIdx.x] += red[threadIdx.x + t]; __syncthreads(); }
  const float inv = rsqrtf(red[0] / Wd + 1e-5f);
  for (int c = threadIdx.x; c < Wd; c += 256) {
    float x = (row[c] - mu) * inv * g[c] + b[c];
    float y = gelu_exact(x);
    out[(size_t)i * Wd + c] = y;
    if (outb) outb[(size_t)i * Wd + c] = f2bf(y);
  }
}

// ---------- stage-2 rel = pos[i] - pos[nb] (bf16, width RPAD, zero padded) ----------
__global__ void rel_kernel(const float* __restrict__ pos, const int* __restrict__ nb2,
                           int base, unsigned short* __restrict__ out){
  int t = blockIdx.x * blockDim.x + threadIdx.x;
  if (t >= E2 * RPAD) return;
  int le = t / RPAD, d = t % RPAD;
  float v = 0.f;
  if (d < 3) {
    int li = le / NBR2, j = le % NBR2;
    int i = base + li;
    int nj = nb2[(size_t)i * NBR2 + j];
    v = pos[(size_t)i * 3 + d] - pos[(size_t)nj * 3 + d];
  }
  out[t] = f2bf(v);
}

// ---------- stage-2 attention-MLP input = a_dst[i] - a_src[nb] + delta (bf16) ----------
__global__ void attn_in_kernel(const float* __restrict__ adst, const float* __restrict__ asrc,
                               const float* __restrict__ delta, const int* __restrict__ nb2,
                               int base, unsigned short* __restrict__ out){
  long long t = (long long)blockIdx.x * blockDim.x + threadIdx.x;
  if (t >= (long long)E2 * H2D) return;
  int le = (int)(t / H2D), c = (int)(t % H2D);
  int li = le / NBR2, j = le % NBR2;
  int i = base + li;
  int nj = nb2[(size_t)i * NBR2 + j];
  float v = adst[(size_t)i * H2D + c] - asrc[(size_t)nj * H2D + c] + delta[(size_t)le * H2D + c];
  out[t] = f2bf(v);
}

// ---------- per-channel softmax over neighbors + weighted sum -> h2 ----------
__global__ void __launch_bounds__(256)
attn_h2_kernel(const float* __restrict__ alphaL, const float* __restrict__ delta,
               const float* __restrict__ v, const int* __restrict__ nb2,
               int base, float* __restrict__ h2){
  const int li = blockIdx.x;
  const int i = base + li;
  const int c = threadIdx.x;          // 256 channels
  float mx = -__int_as_float(0x7F800000);
  for (int j = 0; j < NBR2; ++j)
    mx = fmaxf(mx, alphaL[((size_t)li * NBR2 + j) * H2D + c]);
  float se = 0.f;
  for (int j = 0; j < NBR2; ++j)
    se += __expf(alphaL[((size_t)li * NBR2 + j) * H2D + c] - mx);
  float acc = 0.f;
  for (int j = 0; j < NBR2; ++j) {
    int nj = nb2[(size_t)i * NBR2 + j];
    float w = __expf(alphaL[((size_t)li * NBR2 + j) * H2D + c] - mx) / se;
    acc += w * (v[(size_t)nj * H2D + c] + delta[((size_t)li * NBR2 + j) * H2D + c]);
  }
  h2[(size_t)i * H2D + c] = acc;
}

// ---------- s = h2g@gn_s_w+b (4), hf = h2g@gn_h_w+b (16) ----------
__global__ void shf_kernel(const float* __restrict__ h2g,
                           const float* __restrict__ sw, const float* __restrict__ sb,
                           const float* __restrict__ hw, const float* __restrict__ hb,
                           float* __restrict__ s, float* __restrict__ hf){
  int t = blockIdx.x * blockDim.x + threadIdx.x;
  if (t >= NPTS * 20) return;
  int i = t / 20, o = t % 20;
  const float* row = h2g + (size_t)i * H2D;
  if (o < 4) {
    float acc = sb[o];
    for (int c = 0; c < H2D; ++c) acc += row[c] * sw[(size_t)c * 4 + o];
    s[(size_t)i * 4 + o] = acc;
  } else {
    int oo = o - 4;
    float acc = hb[oo];
    for (int c = 0; c < H2D; ++c) acc += row[c] * hw[(size_t)c * 16 + oo];
    hf[(size_t)i * 16 + oo] = acc;
  }
}

// ---------- stage-3 Gaussian message mean/max aggregation ----------
__global__ void agg_kernel(const float* __restrict__ s, const float* __restrict__ hf,
                           const int* __restrict__ nb3, float* __restrict__ agg){
  int t = blockIdx.x * blockDim.x + threadIdx.x;
  if (t >= NPTS * 16) return;
  int i = t / 16, c = t % 16;
  float s0 = s[(size_t)i * 4 + 0], s1 = s[(size_t)i * 4 + 1];
  float s2 = s[(size_t)i * 4 + 2], s3 = s[(size_t)i * 4 + 3];
  float sum = 0.f, mx = -__int_as_float(0x7F800000);
  for (int j = 0; j < KN3; ++j) {
    int nj = nb3[(size_t)i * KN3 + j];
    float d0 = s0 - s[(size_t)nj * 4 + 0];
    float d1 = s1 - s[(size_t)nj * 4 + 1];
    float d2 = s2 - s[(size_t)nj * 4 + 2];
    float d3 = s3 - s[(size_t)nj * 4 + 3];
    float w = __expf(-10.f * (d0*d0 + d1*d1 + d2*d2 + d3*d3));
    float m = hf[(size_t)nj * 16 + c] * w;
    sum += m; mx = fmaxf(mx, m);
  }
  agg[(size_t)i * 32 + c]      = sum / (float)KN3;
  agg[(size_t)i * 32 + 16 + c] = mx;
}

// ---------- concat [h2g | agg] -> bf16 (288) ----------
__global__ void concat_gn_kernel(const float* __restrict__ h2g, const float* __restrict__ agg,
                                 unsigned short* __restrict__ out){
  long long t = (long long)blockIdx.x * blockDim.x + threadIdx.x;
  if (t >= (long long)NPTS * 288) return;
  int i = (int)(t / 288), c = (int)(t % 288);
  float v = (c < H2D) ? h2g[(size_t)i * H2D + c] : agg[(size_t)i * 32 + (c - H2D)];
  out[t] = f2bf(v);
}

// ---------- concat [h1g | h2g | h3g] -> bf16 (896) ----------
__global__ void concat_ms_kernel(const float* __restrict__ h1g, const float* __restrict__ h2g,
                                 const float* __restrict__ h3g, unsigned short* __restrict__ out){
  long long t = (long long)blockIdx.x * blockDim.x + threadIdx.x;
  if (t >= (long long)NPTS * 896) return;
  int i = (int)(t / 896), c = (int)(t % 896);
  float v;
  if (c < H1D)            v = h1g[(size_t)i * H1D + c];
  else if (c < H1D + H2D) v = h2g[(size_t)i * H2D + (c - H1D)];
  else                    v = h3g[(size_t)i * H3D + (c - H1D - H2D)];
  out[t] = f2bf(v);
}

// ---------- host-side launch helpers ----------
static inline void launch_gemm(const unsigned short* A, const unsigned short* Wp, const float* bias,
                               float* Cf, unsigned short* Cb,
                               int M, int N, int Kp, int act, hipStream_t st){
  dim3 grid((unsigned)((M / 16 + 3) / 4), (unsigned)(N / 64));
  wmma_gemm_kernel<<<grid, dim3(128), 0, st>>>(A, Wp, bias, Cf, Cb, M, N, Kp, act);
}
static inline void pack_w(const float* W, unsigned short* Wp, int N, int K, int Kp, hipStream_t st){
  pack_w_kernel<<<(N * Kp + 255) / 256, 256, 0, st>>>(W, Wp, N, K, Kp);
}

// input index map (setup_inputs dict order: x, pos, params{...})
enum {
  IN_X = 0, IN_POS,
  P_DE_W1, P_DE_B1, P_DE_W2, P_DE_B2, P_DE_W3, P_DE_B3,
  P_LN1_G, P_LN1_B,
  P_PT_LIN, P_PT_SRC, P_PT_DST,
  P_POS_W1, P_POS_B1, P_POS_W2, P_POS_B2,
  P_ATTN_W1, P_ATTN_B1, P_ATTN_W2, P_ATTN_B2,
  P_LN2_G, P_LN2_B,
  P_GN_S_W, P_GN_S_B, P_GN_H_W, P_GN_H_B,
  P_GN_O_W, P_GN_O_B,
  P_LN3_G, P_LN3_B,
  P_FP_W, P_FP_B,
  P_LNF_G, P_LNF_B
};

extern "C" void kernel_launch(void* const* d_in, const int* in_sizes, int n_in,
                              void* d_out, int out_size, void* d_ws, size_t ws_size,
                              hipStream_t stream)
{
  (void)in_sizes; (void)n_in; (void)out_size; (void)ws_size;
  const float* x   = (const float*)d_in[IN_X];
  const float* pos = (const float*)d_in[IN_POS];
  auto P = [&](int idx) -> const float* { return (const float*)d_in[idx]; };

  // ---- workspace bump allocator ----
  char* wsb = (char*)d_ws;
  size_t off = 0;
  auto alloc = [&](size_t bytes) -> void* {
    void* p = wsb + off;
    off = (off + bytes + 255) & ~(size_t)255;
    return p;
  };

  // persistent buffers
  float* xp    = (float*)alloc((size_t)NPTS * DXP * 4);
  int*   nb1   = (int*)  alloc((size_t)NPTS * KN1 * 4);
  int*   nb2   = (int*)  alloc((size_t)NPTS * NBR2 * 4);
  int*   nb3   = (int*)  alloc((size_t)NPTS * KN3 * 4);
  float* h1    = (float*)alloc((size_t)NPTS * H1D * 4);
  float* h1g   = (float*)alloc((size_t)NPTS * H1D * 4);
  unsigned short* h1gb = (unsigned short*)alloc((size_t)NPTS * H1D * 2);
  float* vbuf  = (float*)alloc((size_t)NPTS * H2D * 4);
  float* asrc  = (float*)alloc((size_t)NPTS * H2D * 4);
  float* adst  = (float*)alloc((size_t)NPTS * H2D * 4);
  float* h2    = (float*)alloc((size_t)NPTS * H2D * 4);
  float* h2g   = (float*)alloc((size_t)NPTS * H2D * 4);
  float* sfeat = (float*)alloc((size_t)NPTS * 4 * 4);
  float* hfeat = (float*)alloc((size_t)NPTS * 16 * 4);
  float* agg   = (float*)alloc((size_t)NPTS * 32 * 4);
  float* h3g   = (float*)alloc((size_t)NPTS * H3D * 4);

  // packed bf16 weights (fragment layout), sizes = N * Kp
  unsigned short* w_de1   = (unsigned short*)alloc((size_t)64   * 64  * 2);
  unsigned short* w_de2   = (unsigned short*)alloc((size_t)128  * 64  * 2);
  unsigned short* w_de3   = (unsigned short*)alloc((size_t)128  * 128 * 2);
  unsigned short* w_ptlin = (unsigned short*)alloc((size_t)256  * 128 * 2);
  unsigned short* w_ptsrc = (unsigned short*)alloc((size_t)256  * 128 * 2);
  unsigned short* w_ptdst = (unsigned short*)alloc((size_t)256  * 128 * 2);
  unsigned short* w_pw1   = (unsigned short*)alloc((size_t)64   * 32  * 2);
  unsigned short* w_pw2   = (unsigned short*)alloc((size_t)256  * 64  * 2);
  unsigned short* w_aw1   = (unsigned short*)alloc((size_t)64   * 256 * 2);
  unsigned short* w_aw2   = (unsigned short*)alloc((size_t)256  * 64  * 2);
  unsigned short* w_gno   = (unsigned short*)alloc((size_t)512  * 288 * 2);
  unsigned short* w_fp    = (unsigned short*)alloc((size_t)1024 * 896 * 2);

  const size_t arena = off;   // everything past here is per-phase scratch

  // weight packing (fused f32->bf16 + fragment layout + K zero-pad)
  pack_w(P(P_DE_W1),   w_de1,   64,   38,  64,  stream);
  pack_w(P(P_DE_W2),   w_de2,   128,  64,  64,  stream);
  pack_w(P(P_DE_W3),   w_de3,   128,  128, 128, stream);
  pack_w(P(P_PT_LIN),  w_ptlin, 256,  128, 128, stream);
  pack_w(P(P_PT_SRC),  w_ptsrc, 256,  128, 128, stream);
  pack_w(P(P_PT_DST),  w_ptdst, 256,  128, 128, stream);
  pack_w(P(P_POS_W1),  w_pw1,   64,   3,   32,  stream);
  pack_w(P(P_POS_W2),  w_pw2,   256,  64,  64,  stream);
  pack_w(P(P_ATTN_W1), w_aw1,   64,   256, 256, stream);
  pack_w(P(P_ATTN_W2), w_aw2,   256,  64,  64,  stream);
  pack_w(P(P_GN_O_W),  w_gno,   512,  288, 288, stream);
  pack_w(P(P_FP_W),    w_fp,    1024, 896, 896, stream);

  // ================= Stage 1: edge MLP =================
  build_xp_kernel<<<(NPTS * DXP + 255) / 256, 256, 0, stream>>>(x, pos, xp);
  knn_kernel<<<NPTS, 256, 0, stream>>>(xp, DXP, KN1, KN1, 0, 0, nb1);

  off = arena;
  unsigned short* ebuf   = (unsigned short*)alloc((size_t)E1 * EPAD * 2);
  unsigned short* m64b   = (unsigned short*)alloc((size_t)E1 * 64 * 2);
  unsigned short* m128b  = (unsigned short*)alloc((size_t)E1 * 128 * 2);
  float*          m128f2 = (float*)alloc((size_t)E1 * 128 * 4);

  build_edge_kernel<<<(unsigned)(((long long)E1 * EPAD + 255) / 256), 256, 0, stream>>>(xp, nb1, ebuf);
  launch_gemm(ebuf,  w_de1, P(P_DE_B1), nullptr, m64b,   E1, 64,  64,  1, stream);
  launch_gemm(m64b,  w_de2, P(P_DE_B2), nullptr, m128b,  E1, 128, 64,  1, stream);
  launch_gemm(m128b, w_de3, P(P_DE_B3), m128f2,  nullptr, E1, 128, 128, 0, stream);
  maxpool_kernel<<<(NPTS * H1D + 255) / 256, 256, 0, stream>>>(m128f2, h1);
  ln_gelu_kernel<<<NPTS, 256, 0, stream>>>(h1, P(P_LN1_G), P(P_LN1_B), h1g, h1gb, H1D);

  // ================= Stage 2: point-transformer attention =================
  knn_kernel<<<NPTS, 256, 0, stream>>>(pos, 3, KN2, NBR2, 1, 1, nb2);
  launch_gemm(h1gb, w_ptlin, nullptr, vbuf, nullptr, NPTS, H2D, H1D, 0, stream);
  launch_gemm(h1gb, w_ptsrc, nullptr, asrc, nullptr, NPTS, H2D, H1D, 0, stream);
  launch_gemm(h1gb, w_ptdst, nullptr, adst, nullptr, NPTS, H2D, H1D, 0, stream);

  off = arena;
  unsigned short* relb  = (unsigned short*)alloc((size_t)E2 * RPAD * 2);
  unsigned short* hpb   = (unsigned short*)alloc((size_t)E2 * 64 * 2);
  float*          delta = (float*)alloc((size_t)E2 * H2D * 4);
  unsigned short* ainb  = (unsigned short*)alloc((size_t)E2 * H2D * 2);
  unsigned short* hqb   = (unsigned short*)alloc((size_t)E2 * 64 * 2);
  float*          alph  = (float*)alloc((size_t)E2 * H2D * 4);

  for (int base = 0; base < NPTS; base += CHNK) {
    rel_kernel<<<(E2 * RPAD + 255) / 256, 256, 0, stream>>>(pos, nb2, base, relb);
    launch_gemm(relb, w_pw1, P(P_POS_B1), nullptr, hpb, E2, 64, RPAD, 1, stream);
    launch_gemm(hpb,  w_pw2, P(P_POS_B2), delta, nullptr, E2, H2D, 64, 0, stream);
    attn_in_kernel<<<(unsigned)(((long long)E2 * H2D + 255) / 256), 256, 0, stream>>>(
        adst, asrc, delta, nb2, base, ainb);
    launch_gemm(ainb, w_aw1, P(P_ATTN_B1), nullptr, hqb, E2, 64, H2D, 1, stream);
    launch_gemm(hqb,  w_aw2, P(P_ATTN_B2), alph, nullptr, E2, H2D, 64, 0, stream);
    attn_h2_kernel<<<CHNK, H2D, 0, stream>>>(alph, delta, vbuf, nb2, base, h2);
  }
  ln_gelu_kernel<<<NPTS, 256, 0, stream>>>(h2, P(P_LN2_G), P(P_LN2_B), h2g, nullptr, H2D);

  // ================= Stage 3: Gaussian graph aggregation =================
  shf_kernel<<<(NPTS * 20 + 255) / 256, 256, 0, stream>>>(
      h2g, P(P_GN_S_W), P(P_GN_S_B), P(P_GN_H_W), P(P_GN_H_B), sfeat, hfeat);
  knn_kernel<<<NPTS, 256, 0, stream>>>(sfeat, 4, KN3, KN3, 0, 0, nb3);
  agg_kernel<<<(NPTS * 16 + 255) / 256, 256, 0, stream>>>(sfeat, hfeat, nb3, agg);

  off = arena;
  unsigned short* gnin = (unsigned short*)alloc((size_t)NPTS * 288 * 2);
  float*          t3   = (float*)alloc((size_t)NPTS * H3D * 4);
  concat_gn_kernel<<<(unsigned)(((long long)NPTS * 288 + 255) / 256), 256, 0, stream>>>(h2g, agg, gnin);
  launch_gemm(gnin, w_gno, P(P_GN_O_B), t3, nullptr, NPTS, H3D, 288, 0, stream);
  ln_gelu_kernel<<<NPTS, 256, 0, stream>>>(t3, P(P_LN3_G), P(P_LN3_B), h3g, nullptr, H3D);

  // ================= Final fusion =================
  off = arena;
  unsigned short* msb = (unsigned short*)alloc((size_t)NPTS * 896 * 2);
  float*          tf  = (float*)alloc((size_t)NPTS * OUTD * 4);
  concat_ms_kernel<<<(unsigned)(((long long)NPTS * 896 + 255) / 256), 256, 0, stream>>>(h1g, h2g, h3g, msb);
  launch_gemm(msb, w_fp, P(P_FP_B), tf, nullptr, NPTS, OUTD, 896, 0, stream);
  ln_gelu_kernel<<<NPTS, 256, 0, stream>>>(tf, P(P_LNF_G), P(P_LNF_B), (float*)d_out, nullptr, OUTD);
}
